// SetAbstraction_7765300871439
// MI455X (gfx1250) — matrix operand order repaired and checked by reading, not compiled, not run on hardware
//
#include <hip/hip_runtime.h>

typedef __attribute__((ext_vector_type(16))) _Float16 v16h;
typedef __attribute__((ext_vector_type(8)))  float    v8f;

#define B_  8
#define N_  4096
#define M_  1024
#define K_  32
#define CF_ 64
#define R_  (B_*M_*K_)      // 262144 rows (points)
#define NB_ 256             // stats partial blocks
#define RAD2 0.04f          // radius^2

// ---------------------------------------------------------------------------
// 1) Ball query: one wave32 per query; emit first-K in-ball indices in
//    ascending index order via ballot + prefix popcount; pad with first/0.
// ---------------------------------------------------------------------------
__global__ __launch_bounds__(256) void ball_query_kernel(
    const float* __restrict__ p, const float* __restrict__ q, int* __restrict__ idx)
{
  const int lane = threadIdx.x & 31;
  const int qid  = (blockIdx.x * 256 + threadIdx.x) >> 5;   // 0..B*M-1
  const int b = qid / M_;
  const int m = qid % M_;
  const float qx = q[(qid)*3 + 0];
  const float qy = q[(qid)*3 + 1];
  const float qz = q[(qid)*3 + 2];
  (void)m;
  int* orow = idx + (long)qid * K_;
  int cnt = 0, first = 0;
  for (int base = 0; base < N_; base += 32) {
    const int i = base + lane;
    const float* pr = p + ((long)b*N_ + i)*3;
    float dx = pr[0]-qx, dy = pr[1]-qy, dz = pr[2]-qz;
    float d2 = dx*dx + dy*dy + dz*dz;
    bool inb = d2 < RAD2;
    unsigned mask = __builtin_amdgcn_ballot_w32(inb);
    if (cnt == 0 && mask != 0u) first = base + __builtin_ctz(mask);
    if (inb) {
      int slot = cnt + __popc(mask & ((1u << lane) - 1u));
      if (slot < K_) orow[slot] = i;
    }
    cnt += __popc(mask);
    if (cnt >= K_) break;
  }
  for (int s = cnt + lane; s < K_; s += 32) orow[s] = (cnt > 0) ? first : 0;
}

// ---------------------------------------------------------------------------
// 2) Transpose x: (B,64,N) -> (B,N,64) so per-point feature gather is one
//    contiguous 256B row. LDS-tiled, coalesced both ways.
// ---------------------------------------------------------------------------
__global__ __launch_bounds__(256) void transpose_x_kernel(
    const float* __restrict__ x, float* __restrict__ xT)
{
  __shared__ float tile[64][33];
  const int t  = threadIdx.x;
  const int b  = blockIdx.x / (N_/32);
  const int n0 = (blockIdx.x % (N_/32)) * 32;
  #pragma unroll
  for (int i = 0; i < 8; i++) {
    int e = t + 256*i;
    int ch = e >> 5, n = e & 31;
    tile[ch][n] = x[((long)b*CF_ + ch)*N_ + n0 + n];
  }
  __syncthreads();
  #pragma unroll
  for (int i = 0; i < 8; i++) {
    int e = t + 256*i;
    int n = e >> 6, ch = e & 63;
    xT[((long)b*N_ + n0 + n)*CF_ + ch] = tile[ch][n];
  }
}

// ---------------------------------------------------------------------------
// 3) Layer-0: fused gather(pg-q, xg) + GEMM with W0 (64x67), K padded to 96.
//    Wave computes 8 strips of 16 rows x Cout=64 using v_wmma_f32_16x16x32_f16.
//    A layout (16-bit 16x32): lanes 0-15 row M=lane, K={0..7,16..23};
//    lanes 16-31 same rows, K={8..15,24..31}.
//    B layout (32x16):       lanes 0-15 N=lane,  K={0..7,16..23};
//    lanes 16-31 N=lane-16,  K={8..15,24..31}.
//    D layout (f32 16x16):   lane<16: elem v -> (M=v,N=lane); lane>=16: (M=8+v,N=lane-16).
// ---------------------------------------------------------------------------
__global__ __launch_bounds__(256) void gemm0_kernel(
    const float* __restrict__ p, const float* __restrict__ q,
    const float* __restrict__ xT, const int* __restrict__ idx,
    const float* __restrict__ W, const float* __restrict__ bias,
    float* __restrict__ hout)
{
  constexpr int CIN = 67, NCH = 3, NT = 4, COUT = 64;
  const int lane = threadIdx.x & 31;
  const int wave = threadIdx.x >> 5;
  const int col  = lane & 15;
  const int hsel = lane >> 4;

  v16h Bf[NT][NCH];
  #pragma unroll
  for (int n = 0; n < NT; n++) {
    const float* wr = W + (n*16 + col)*CIN;
    #pragma unroll
    for (int j = 0; j < NCH; j++) {
      const int kbase = 32*j + hsel*8;
      v16h f;
      #pragma unroll
      for (int e = 0; e < 16; e++) {
        int k = kbase + e + (e & 8);
        f[e] = (_Float16)((k < CIN) ? wr[k] : 0.f);
      }
      Bf[n][j] = f;
    }
  }
  float bv[NT];
  #pragma unroll
  for (int n = 0; n < NT; n++) bv[n] = bias[n*16 + col];

  const long rowWaveBase = (long)blockIdx.x*1024 + (long)wave*128;
  for (int s = 0; s < 8; s++) {
    const long row = rowWaveBase + s*16 + col;
    const int b = (int)(row / ((long)M_*K_));
    const int m = (int)((row / K_) % M_);
    const int i = idx[row];
    const float* xr = xT + ((long)b*N_ + i)*CF_;
    const float* pr = p  + ((long)b*N_ + i)*3;
    const float* qr = q  + ((long)b*M_ + m)*3;
    v16h Af[NCH];
    #pragma unroll
    for (int j = 0; j < NCH; j++) {
      const int cbase = 32*j + hsel*8;
      v16h f;
      #pragma unroll
      for (int e = 0; e < 8; e++) {
        int c1 = cbase + e;
        int c2 = c1 + 16;
        float v1 = (c1 >= CIN) ? 0.f : ((c1 >= 3) ? xr[c1-3] : (pr[c1]-qr[c1]));
        float v2 = (c2 >= CIN) ? 0.f : ((c2 >= 3) ? xr[c2-3] : (pr[c2]-qr[c2]));
        f[e]   = (_Float16)v1;
        f[e+8] = (_Float16)v2;
      }
      Af[j] = f;
    }
    float* orow = hout + (rowWaveBase + s*16)*COUT;
    #pragma unroll
    for (int n = 0; n < NT; n++) {
      v8f acc;
      #pragma unroll
      for (int e = 0; e < 8; e++) acc[e] = bv[n];
      #pragma unroll
      for (int j = 0; j < NCH; j++)
        acc = __builtin_amdgcn_wmma_f32_16x16x32_f16(false, Af[j], false, Bf[n][j],
                                                     (short)0, acc, false, false);
      #pragma unroll
      for (int v = 0; v < 8; v++)
        orow[(long)(hsel*8 + v)*COUT + n*16 + col] = acc[v];
    }
  }
}

// ---------------------------------------------------------------------------
// 4) Layers 1/2: BN(scale,shift)+ReLU on input load, f16 convert, WMMA GEMM.
// ---------------------------------------------------------------------------
template<int CIN, int COUT>
__global__ __launch_bounds__(256) void gemm_bn_kernel(
    const float* __restrict__ hin, const float* __restrict__ scale,
    const float* __restrict__ shift, const float* __restrict__ W,
    const float* __restrict__ bias, float* __restrict__ hout)
{
  constexpr int NCH = CIN / 32;
  constexpr int NT  = COUT / 16;
  const int lane = threadIdx.x & 31;
  const int wave = threadIdx.x >> 5;
  const int col  = lane & 15;
  const int hsel = lane >> 4;

  v16h Bf[NT][NCH];
  #pragma unroll
  for (int n = 0; n < NT; n++) {
    const float* wr = W + (n*16 + col)*CIN;
    #pragma unroll
    for (int j = 0; j < NCH; j++) {
      const int kbase = 32*j + hsel*8;
      v16h f;
      #pragma unroll
      for (int e = 0; e < 16; e++) f[e] = (_Float16)wr[kbase + e + (e & 8)];
      Bf[n][j] = f;
    }
  }
  float bv[NT];
  #pragma unroll
  for (int n = 0; n < NT; n++) bv[n] = bias[n*16 + col];

  float scv[NCH*16], shv[NCH*16];
  #pragma unroll
  for (int j = 0; j < NCH; j++)
    #pragma unroll
    for (int e = 0; e < 8; e++) {
      int c1 = 32*j + hsel*8 + e;
      scv[j*16+e]   = scale[c1];      shv[j*16+e]   = shift[c1];
      scv[j*16+8+e] = scale[c1+16];   shv[j*16+8+e] = shift[c1+16];
    }

  const long rowWaveBase = (long)blockIdx.x*1024 + (long)wave*128;
  for (int s = 0; s < 8; s++) {
    const float* hr = hin + (rowWaveBase + s*16 + col)*CIN;
    v16h Af[NCH];
    #pragma unroll
    for (int j = 0; j < NCH; j++) {
      const int cbase = 32*j + hsel*8;
      v16h f;
      #pragma unroll
      for (int e = 0; e < 8; e++) {
        float x1 = hr[cbase + e];
        float x2 = hr[cbase + e + 16];
        f[e]   = (_Float16)fmaxf(fmaf(x1, scv[j*16+e],   shv[j*16+e]),   0.f);
        f[e+8] = (_Float16)fmaxf(fmaf(x2, scv[j*16+8+e], shv[j*16+8+e]), 0.f);
      }
      Af[j] = f;
    }
    float* orow = hout + (rowWaveBase + s*16)*COUT;
    #pragma unroll
    for (int n = 0; n < NT; n++) {
      v8f acc;
      #pragma unroll
      for (int e = 0; e < 8; e++) acc[e] = bv[n];
      #pragma unroll
      for (int j = 0; j < NCH; j++)
        acc = __builtin_amdgcn_wmma_f32_16x16x32_f16(false, Af[j], false, Bf[n][j],
                                                     (short)0, acc, false, false);
      #pragma unroll
      for (int v = 0; v < 8; v++)
        orow[(long)(hsel*8 + v)*COUT + n*16 + col] = acc[v];
    }
  }
}

// ---------------------------------------------------------------------------
// 5) Deterministic BN stats: stage-1 per-block partial sum/sumsq per channel,
//    stage-2 fixed-order reduce -> fused scale/shift.
// ---------------------------------------------------------------------------
template<int C>
__global__ __launch_bounds__(256) void stats_partial_kernel(
    const float* __restrict__ h, float* __restrict__ psum, float* __restrict__ psq)
{
  constexpr int GR  = 256 / C;
  constexpr int RPB = R_ / NB_;
  const int t  = threadIdx.x;
  const int c  = t % C;
  const int rg = t / C;
  const long base = (long)blockIdx.x * RPB;
  float s = 0.f, sq = 0.f;
  for (int i = rg; i < RPB; i += GR) {
    float v = h[(base + i)*C + c];
    s += v; sq += v*v;
  }
  __shared__ float ls[256], lq[256];
  ls[t] = s; lq[t] = sq;
  __syncthreads();
  if (t < C) {
    float ts = 0.f, tq = 0.f;
    #pragma unroll
    for (int g = 0; g < GR; g++) { ts += ls[g*C + t]; tq += lq[g*C + t]; }
    psum[blockIdx.x*C + t] = ts;
    psq [blockIdx.x*C + t] = tq;
  }
}

template<int C>
__global__ void stats_final_kernel(
    const float* __restrict__ psum, const float* __restrict__ psq,
    const float* __restrict__ g, const float* __restrict__ be,
    float* __restrict__ scale, float* __restrict__ shift)
{
  const int c = threadIdx.x;
  if (c < C) {
    float s = 0.f, sq = 0.f;
    for (int b = 0; b < NB_; b++) { s += psum[b*C + c]; sq += psq[b*C + c]; }
    const float inv = 1.0f / (float)R_;
    float mu  = s * inv;
    float var = fmaxf(sq * inv - mu*mu, 0.f);
    float rs  = rsqrtf(var + 1e-5f);
    float sc  = rs * g[c];
    scale[c] = sc;
    shift[c] = be[c] - mu * sc;
  }
}

// ---------------------------------------------------------------------------
// 6) Finalize: BN+ReLU on h2, max over K, LDS transpose -> out (B,128,M).
// ---------------------------------------------------------------------------
__global__ __launch_bounds__(256) void finalize_kernel(
    const float* __restrict__ h2, const float* __restrict__ scale,
    const float* __restrict__ shift, float* __restrict__ out)
{
  __shared__ float tile[32][129];
  const int t  = threadIdx.x;
  const int b  = blockIdx.x / (M_/32);
  const int m0 = (blockIdx.x % (M_/32)) * 32;
  const int ch = t % 128;
  const float sc = scale[ch], sh = shift[ch];
  for (int mm = t/128; mm < 32; mm += 2) {
    const long rb = ((long)b*M_ + m0 + mm) * K_;
    float mx = 0.f;   // ReLU outputs are >= 0
    #pragma unroll 4
    for (int k = 0; k < K_; k++) {
      float v = h2[(rb + k)*128 + ch];
      v = fmaxf(fmaf(v, sc, sh), 0.f);
      mx = fmaxf(mx, v);
    }
    tile[mm][ch] = mx;
  }
  __syncthreads();
  for (int cg = 0; cg < 128; cg += 8) {
    int ch2 = cg + t/32;
    int j   = t & 31;
    out[((long)b*128 + ch2)*M_ + m0 + j] = tile[j][ch2];
  }
}

__global__ void copy_q_kernel(const float* __restrict__ q, float* __restrict__ out)
{
  int i = blockIdx.x*256 + threadIdx.x;
  if (i < B_*M_*3) out[i] = q[i];
}

// ---------------------------------------------------------------------------
extern "C" void kernel_launch(void* const* d_in, const int* in_sizes, int n_in,
                              void* d_out, int out_size, void* d_ws, size_t ws_size,
                              hipStream_t stream)
{
  (void)in_sizes; (void)n_in; (void)out_size; (void)ws_size;
  const float* p   = (const float*)d_in[0];
  const float* q   = (const float*)d_in[1];
  const float* x   = (const float*)d_in[2];
  const float* W0  = (const float*)d_in[3];
  const float* b0  = (const float*)d_in[4];
  const float* g0  = (const float*)d_in[5];
  const float* be0 = (const float*)d_in[6];
  const float* W1  = (const float*)d_in[7];
  const float* b1  = (const float*)d_in[8];
  const float* g1  = (const float*)d_in[9];
  const float* be1 = (const float*)d_in[10];
  const float* W2  = (const float*)d_in[11];
  const float* b2  = (const float*)d_in[12];
  const float* g2  = (const float*)d_in[13];
  const float* be2 = (const float*)d_in[14];

  char* ws = (char*)d_ws;
  int*   idx  = (int*)ws;                                   // 1 MB
  float* xT   = (float*)(ws + ((size_t)1  << 20));          // 8 MB
  float* psum = (float*)(ws + ((size_t)9  << 20));          // 128 KB
  float* psq  = psum + NB_*128;                             // 128 KB
  float* sc0  = psq  + NB_*128;
  float* sh0  = sc0 + 64;
  float* sc1  = sh0 + 64;
  float* sh1  = sc1 + 64;
  float* sc2  = sh1 + 64;
  float* sh2  = sc2 + 128;
  float* h1   = (float*)(ws + ((size_t)10 << 20));          // 64 MB
  float* hbig = (float*)(ws + ((size_t)74 << 20));          // 128 MB (h0 then h2)
  float* h0 = hbig;
  float* h2 = hbig;
  float* outq = (float*)d_out;
  float* outf = (float*)d_out + (size_t)B_*M_*3;

  ball_query_kernel<<<(B_*M_)/8, 256, 0, stream>>>(p, q, idx);
  transpose_x_kernel<<<B_*(N_/32), 256, 0, stream>>>(x, xT);

  gemm0_kernel<<<R_/1024, 256, 0, stream>>>(p, q, xT, idx, W0, b0, h0);
  stats_partial_kernel<64><<<NB_, 256, 0, stream>>>(h0, psum, psq);
  stats_final_kernel<64><<<1, 64, 0, stream>>>(psum, psq, g0, be0, sc0, sh0);

  gemm_bn_kernel<64,64><<<R_/1024, 256, 0, stream>>>(h0, sc0, sh0, W1, b1, h1);
  stats_partial_kernel<64><<<NB_, 256, 0, stream>>>(h1, psum, psq);
  stats_final_kernel<64><<<1, 64, 0, stream>>>(psum, psq, g1, be1, sc1, sh1);

  gemm_bn_kernel<64,128><<<R_/1024, 256, 0, stream>>>(h1, sc1, sh1, W2, b2, h2);
  stats_partial_kernel<128><<<NB_, 256, 0, stream>>>(h2, psum, psq);
  stats_final_kernel<128><<<1, 128, 0, stream>>>(psum, psq, g2, be2, sc2, sh2);

  copy_q_kernel<<<(B_*M_*3 + 255)/256, 256, 0, stream>>>(q, outq);
  finalize_kernel<<<B_*(M_/32), 256, 0, stream>>>(h2, sc2, sh2, outf);
}
